// LSTMBased_79671643341697
// MI455X (gfx1250) — matrix operand order repaired and checked by reading, not compiled, not run on hardware
//
#include <hip/hip_runtime.h>

// ===========================================================================
// LSTM seq2seq (2-layer biLSTM encoder + attention decoder) for MI455X gfx1250
// All GEMMs use V_WMMA_F32_16X16X32_BF16 (bf16 A/B, fp32 accumulate).
// Batch (8) is padded to 16 to fill the WMMA M tile; pad rows carry garbage
// that is never stored to the output.
// GEMM kernels process 4 N-tiles per wave so the A fragment is loaded once
// per k-step and reused across 4 WMMAs.
// Workspace layout uses ~278 MB of d_ws.
// ===========================================================================

typedef __attribute__((ext_vector_type(16))) __bf16 v16bf;
typedef __attribute__((ext_vector_type(8)))  float  v8f;

union BF16Frag { unsigned int w[8]; unsigned short u[16]; v16bf v; };

__device__ __forceinline__ unsigned short f2bf(float f) {
  unsigned int u = __float_as_uint(f);
  unsigned int r = (u + 0x7FFFu + ((u >> 16) & 1u)) >> 16;   // RNE
  return (unsigned short)r;
}
__device__ __forceinline__ float bf2f(unsigned short u) {
  return __uint_as_float(((unsigned int)u) << 16);
}
__device__ __forceinline__ float sigm(float x) { return 1.0f / (1.0f + __expf(-x)); }

// ---- WMMA fragment helpers (layouts per CDNA5 ISA 7.12.2) ------------------
// A: 16x32 bf16, row-major source, rows m0..m0+15, cols k0..k0+31
__device__ __forceinline__ v16bf load_a16(const unsigned short* A, int lda, int m0, int k0) {
  int lane = threadIdx.x & 31;
  int half = lane >> 4, m = lane & 15;
  const unsigned short* p = A + (size_t)(m0 + m) * lda + k0 + 8 * half;
  BF16Frag f;
#pragma unroll
  for (int v = 0; v < 4; ++v) {
    f.w[v]     = *(const unsigned int*)(p + 2 * v);        // K = 8*half + 2v, +1
    f.w[v + 4] = *(const unsigned int*)(p + 16 + 2 * v);   // K = 16 + 8*half + 2v, +1
  }
  return f.v;
}
// B: 32x16 bf16 where B = W^T, W row-major [N_total][K_total]; cols n0..n0+15
__device__ __forceinline__ v16bf load_b16(const unsigned short* W, int ldw, int n0, int k0) {
  int lane = threadIdx.x & 31;
  int half = lane >> 4, n = lane & 15;
  const unsigned short* p = W + (size_t)(n0 + n) * ldw + k0 + 8 * half;
  BF16Frag f;
#pragma unroll
  for (int v = 0; v < 4; ++v) {
    f.w[v]     = *(const unsigned int*)(p + 2 * v);        // K = 8*half + 2v, +1
    f.w[v + 4] = *(const unsigned int*)(p + 16 + 2 * v);   // K = 16 + 8*half + 2v, +1
  }
  return f.v;
}
__device__ __forceinline__ v8f wmma_bf16(v16bf a, v16bf b, v8f c) {
  return __builtin_amdgcn_wmma_f32_16x16x32_bf16(false, a, false, b, (short)0, c, false, false);
}
__device__ __forceinline__ void store_c(float* C, int ldc, int m0, int n0, v8f acc) {
  int lane = threadIdx.x & 31, half = lane >> 4, n = lane & 15;
#pragma unroll
  for (int v = 0; v < 8; ++v) C[(size_t)(m0 + v + 8 * half) * ldc + n0 + n] = acc[v];
}
__device__ __forceinline__ v8f load_c(const float* C, int ldc, int m0, int n0) {
  int lane = threadIdx.x & 31, half = lane >> 4, n = lane & 15;
  v8f acc;
#pragma unroll
  for (int v = 0; v < 8; ++v) acc[v] = C[(size_t)(m0 + v + 8 * half) * ldc + n0 + n];
  return acc;
}

// ---- utility kernels -------------------------------------------------------
__global__ void k_cvt(const float* __restrict__ s, unsigned short* __restrict__ d, size_t n) {
  size_t i = (size_t)blockIdx.x * blockDim.x + threadIdx.x;
  size_t st = (size_t)gridDim.x * blockDim.x;
  for (; i < n; i += st) d[i] = f2bf(s[i]);
}
// packed decoder weight [4096][2304] = [Wih (k<1280) | Whh (k>=1280)]
__global__ void k_pack_wdec(const float* __restrict__ Wih, const float* __restrict__ Whh,
                            unsigned short* __restrict__ Wd) {
  size_t i = (size_t)blockIdx.x * blockDim.x + threadIdx.x;
  size_t st = (size_t)gridDim.x * blockDim.x;
  for (; i < 4096ull * 2304; i += st) {
    size_t n = i / 2304, k = i % 2304;
    float v = (k < 1280) ? Wih[n * 1280 + k] : Whh[n * 1024 + (k - 1280)];
    Wd[i] = f2bf(v);
  }
}
__global__ void k_bsum(const float* a, const float* b, float* o, int n) {
  int i = blockIdx.x * blockDim.x + threadIdx.x;
  if (i < n) o[i] = a[i] + b[i];
}
// xs[s][b][e] (b padded to 16), bf16
__global__ void k_embed_x(const float* __restrict__ emb, const int* __restrict__ x,
                          unsigned short* __restrict__ xs) {
  size_t i = (size_t)blockIdx.x * blockDim.x + threadIdx.x;
  if (i >= 512ull * 16 * 256) return;
  int e = (int)(i & 255), b = (int)((i >> 8) & 15), s = (int)(i >> 12);
  float v = 0.f;
  if (b < 8) v = emb[(size_t)x[b * 512 + s] * 256 + e];
  xs[i] = f2bf(v);
}
// decoder inputs: step 0 -> dec_emb[CLS=1], step t>=1 -> dec_emb[y[b,t]]
__global__ void k_embed_y(const float* __restrict__ demb, const int* __restrict__ y,
                          unsigned short* __restrict__ di) {
  size_t i = (size_t)blockIdx.x * blockDim.x + threadIdx.x;
  if (i >= 255ull * 16 * 256) return;
  int e = (int)(i & 255), b = (int)((i >> 8) & 15), t = (int)(i >> 12);
  float v = 0.f;
  if (b < 8) { int tok = (t == 0) ? 1 : y[b * 256 + t]; v = demb[(size_t)tok * 256 + e]; }
  di[i] = f2bf(v);
}
__global__ void k_zero_state(unsigned short* xt, float* h, float* c) {
  int i = blockIdx.x * blockDim.x + threadIdx.x;
  if (i < 16 * 2304) xt[i] = 0;
  if (i < 16 * 1024) { h[i] = 0.f; c[i] = 0.f; }
}

// ---- encoder: loop-invariant input GEMM, both directions -------------------
// G[dir][t*16+b][n] = bih[n]+bhh[n] + sum_k A[t*16+b][k] * W[n][k]
// Each wave: one 16-row M tile x four 16-col N tiles (A fragment reused 4x).
__global__ void k_gate_gemm(const unsigned short* __restrict__ A, int K,
                            const unsigned short* __restrict__ W0,
                            const unsigned short* __restrict__ W1,
                            const float* __restrict__ bi0, const float* __restrict__ bh0,
                            const float* __restrict__ bi1, const float* __restrict__ bh1,
                            float* __restrict__ G) {
  int wave = threadIdx.x >> 5;
  int grp = blockIdx.x * 8 + wave;   // 2 dirs * 512 mtiles * 32 ngroups = 32768
  int dir = grp >> 14;
  int rem = grp & 16383;
  int m0 = (rem >> 5) << 4;
  int nb = (rem & 31) << 6;          // 64-column group
  const unsigned short* W = dir ? W1 : W0;
  const float* bi = dir ? bi1 : bi0;
  const float* bh = dir ? bh1 : bh0;
  int lane = threadIdx.x & 31, n = lane & 15;
  v8f acc[4];
#pragma unroll
  for (int i = 0; i < 4; ++i) {
    float bias = bi[nb + 16 * i + n] + bh[nb + 16 * i + n];
#pragma unroll
    for (int v = 0; v < 8; ++v) acc[i][v] = bias;
  }
  for (int kk = 0; kk < K; kk += 32) {
    v16bf a = load_a16(A, K, m0, kk);
#pragma unroll
    for (int i = 0; i < 4; ++i)
      acc[i] = wmma_bf16(a, load_b16(W, K, nb + 16 * i, kk), acc[i]);
  }
  float* Gd = G + (size_t)dir * 512 * 16 * 2048;
#pragma unroll
  for (int i = 0; i < 4; ++i) store_c(Gd, 2048, m0, nb + 16 * i, acc[i]);
}

// ---- encoder: sequential recurrence, one workgroup per direction -----------
// h kept bf16 in LDS; c kept in registers in C-fragment layout; each wave owns
// the same 16 hidden columns of all 4 gates -> gate math stays in registers.
__global__ void __launch_bounds__(1024) k_lstm_seq(
    const float* __restrict__ G,
    const unsigned short* __restrict__ Whh0,
    const unsigned short* __restrict__ Whh1,
    unsigned short* __restrict__ Y) {            // [S][16][1024], this dir -> cols dir*512..
  __shared__ unsigned short h_sh[16 * 512];
  int dir = blockIdx.x;
  const unsigned short* Whh = dir ? Whh1 : Whh0;
  int tid = threadIdx.x;
  for (int i = tid; i < 16 * 512; i += 1024) h_sh[i] = 0;
  __syncthreads();
  int wave = tid >> 5;
  int lane = tid & 31, half = lane >> 4, n = lane & 15;
  int j0 = wave * 16;                            // 32 waves cover hidden 0..511
  v8f c_frag;
#pragma unroll
  for (int v = 0; v < 8; ++v) c_frag[v] = 0.f;
  for (int step = 0; step < 512; ++step) {
    int t = dir ? (511 - step) : step;
    const float* Gt = G + ((size_t)dir * 512 + t) * 16 * 2048;
    v8f ai = load_c(Gt, 2048, 0, j0);
    v8f af = load_c(Gt, 2048, 0, 512 + j0);
    v8f ag = load_c(Gt, 2048, 0, 1024 + j0);
    v8f ao = load_c(Gt, 2048, 0, 1536 + j0);
    for (int kk = 0; kk < 512; kk += 32) {
      v16bf a = load_a16(h_sh, 512, 0, kk);
      ai = wmma_bf16(a, load_b16(Whh, 512, j0, kk), ai);
      af = wmma_bf16(a, load_b16(Whh, 512, 512 + j0, kk), af);
      ag = wmma_bf16(a, load_b16(Whh, 512, 1024 + j0, kk), ag);
      ao = wmma_bf16(a, load_b16(Whh, 512, 1536 + j0, kk), ao);
    }
    __syncthreads();                             // all reads of old h done
#pragma unroll
    for (int v = 0; v < 8; ++v) {
      float cv = sigm(af[v]) * c_frag[v] + sigm(ai[v]) * tanhf(ag[v]);
      float hv = sigm(ao[v]) * tanhf(cv);
      c_frag[v] = cv;
      int m = v + 8 * half;
      unsigned short hb = f2bf(hv);
      h_sh[m * 512 + j0 + n] = hb;
      Y[((size_t)t * 16 + m) * 1024 + dir * 512 + j0 + n] = hb;
    }
    __syncthreads();                             // new h visible
  }
}

// ---- attention precompute: enc_proj[b][s][u] -------------------------------
__global__ void k_encproj(const unsigned short* __restrict__ enc,  // [512][16][1024]
                          const float* __restrict__ W1,            // [10][1024]
                          float* __restrict__ proj) {              // [8][512][10]
  int wave = threadIdx.x >> 5, lane = threadIdx.x & 31;
  int r = blockIdx.x * 8 + wave;                 // 4096 rows = b*512+s
  int b = r >> 9, s = r & 511;
  const unsigned short* e = enc + ((size_t)s * 16 + b) * 1024;
  for (int u = 0; u < 10; ++u) {
    float acc = 0.f;
    for (int k = lane; k < 1024; k += 32) acc += bf2f(e[k]) * W1[u * 1024 + k];
#pragma unroll
    for (int off = 16; off; off >>= 1) acc += __shfl_xor(acc, off, 32);
    if (lane == 0) proj[(size_t)r * 10 + u] = acc;
  }
}

// ---- decoder per-step: attention + context + din ---------------------------
__global__ void k_att(int t,
                      const float* __restrict__ hdec,         // [16][1024] fp32
                      const float* __restrict__ W2,           // [10][1024]
                      const float* __restrict__ attV,         // [10]
                      const float* __restrict__ proj,         // [8][512][10]
                      const unsigned short* __restrict__ enc, // [512][16][1024]
                      const unsigned short* __restrict__ dins,// [255][16][256]
                      unsigned short* __restrict__ xt) {      // [16][2304]
  __shared__ float red[256];
  __shared__ float w2h[10];
  __shared__ float sc[512];
  int b = blockIdx.x, tid = threadIdx.x;
  for (int u = 0; u < 10; ++u) {                 // w2h = W2 @ h[b]
    float p = 0.f;
    for (int k = tid; k < 1024; k += 256) p += W2[u * 1024 + k] * hdec[b * 1024 + k];
    red[tid] = p; __syncthreads();
    for (int off = 128; off; off >>= 1) { if (tid < off) red[tid] += red[tid + off]; __syncthreads(); }
    if (tid == 0) w2h[u] = red[0];
    __syncthreads();
  }
  const float* pb = proj + (size_t)b * 512 * 10;
  for (int s = tid; s < 512; s += 256) {
    float su = 0.f;
#pragma unroll
    for (int u = 0; u < 10; ++u) su += tanhf(pb[s * 10 + u] + w2h[u]) * attV[u];
    sc[s] = su;
  }
  __syncthreads();
  float m = -3.4e38f;
  for (int s = tid; s < 512; s += 256) m = fmaxf(m, sc[s]);
  red[tid] = m; __syncthreads();
  for (int off = 128; off; off >>= 1) { if (tid < off) red[tid] = fmaxf(red[tid], red[tid + off]); __syncthreads(); }
  float mx = red[0]; __syncthreads();
  float p = 0.f;
  for (int s = tid; s < 512; s += 256) { float e = __expf(sc[s] - mx); sc[s] = e; p += e; }
  red[tid] = p; __syncthreads();
  for (int off = 128; off; off >>= 1) { if (tid < off) red[tid] += red[tid + off]; __syncthreads(); }
  float inv = 1.0f / red[0];
  __syncthreads();
  for (int k = tid; k < 1024; k += 256) {        // ctx -> xt cols [256,1280)
    float acc = 0.f;
    for (int s = 0; s < 512; ++s) acc += sc[s] * bf2f(enc[((size_t)s * 16 + b) * 1024 + k]);
    xt[b * 2304 + 256 + k] = f2bf(acc * inv);
  }
  for (int k = tid; k < 256; k += 256)           // din -> xt cols [0,256)
    xt[b * 2304 + k] = dins[((size_t)t * 16 + b) * 256 + k];
}

// ---- decoder cell GEMM: g = xt @ [Wih|Whh]^T + (bih+bhh) -------------------
// 64 groups of 4 N-tiles; 8 waves/block -> 8 blocks.
__global__ void k_cell_gemm(const unsigned short* __restrict__ xt,  // [16][2304]
                            const unsigned short* __restrict__ Wd,  // [4096][2304]
                            const float* __restrict__ bsum,         // [4096]
                            float* __restrict__ g) {                // [16][4096]
  int wave = threadIdx.x >> 5;
  int grp = blockIdx.x * 8 + wave;               // 64 groups
  int nb = grp << 6;
  int lane = threadIdx.x & 31, n = lane & 15;
  v8f acc[4];
#pragma unroll
  for (int i = 0; i < 4; ++i) {
    float bias = bsum[nb + 16 * i + n];
#pragma unroll
    for (int v = 0; v < 8; ++v) acc[i][v] = bias;
  }
  for (int kk = 0; kk < 2304; kk += 32) {
    v16bf a = load_a16(xt, 2304, 0, kk);
#pragma unroll
    for (int i = 0; i < 4; ++i)
      acc[i] = wmma_bf16(a, load_b16(Wd, 2304, nb + 16 * i, kk), acc[i]);
  }
#pragma unroll
  for (int i = 0; i < 4; ++i) store_c(g, 4096, 0, nb + 16 * i, acc[i]);
}

__global__ void k_cell_act(const float* __restrict__ g, float* __restrict__ h,
                           float* __restrict__ c, unsigned short* __restrict__ xt) {
  int e = blockIdx.x * blockDim.x + threadIdx.x; // 16384 = 16*1024
  int b = e >> 10, j = e & 1023;
  float gi = g[(size_t)b * 4096 + j];
  float gf = g[(size_t)b * 4096 + 1024 + j];
  float gg = g[(size_t)b * 4096 + 2048 + j];
  float go = g[(size_t)b * 4096 + 3072 + j];
  float cv = sigm(gf) * c[e] + sigm(gi) * tanhf(gg);
  float hv = sigm(go) * tanhf(cv);
  c[e] = cv;
  h[e] = hv;
  xt[b * 2304 + 1280 + j] = f2bf(hv);            // h bf16 -> xt cols [1280,2304)
}

// ---- decoder output projection: pred = h @ Wv^T + bv (Wv bf16, L2-resident)
// 500 groups of 4 N-tiles; 8 waves/block -> 63 blocks (last 4 waves idle).
__global__ void k_pred(int t, const unsigned short* __restrict__ xt,
                       const unsigned short* __restrict__ Wv,   // [32000][1024] bf16
                       const float* __restrict__ bv,
                       float* __restrict__ out) {               // [8][255][32000]
  int wave = threadIdx.x >> 5;
  int grp = blockIdx.x * 8 + wave;               // 500 groups of 64 columns
  if (grp >= 500) return;
  int nb = grp << 6;
  int lane = threadIdx.x & 31, half = lane >> 4, n = lane & 15;
  v8f acc[4];
#pragma unroll
  for (int i = 0; i < 4; ++i) {
    float bias = bv[nb + 16 * i + n];
#pragma unroll
    for (int v = 0; v < 8; ++v) acc[i][v] = bias;
  }
  for (int kk = 0; kk < 1024; kk += 32) {
    v16bf a = load_a16(xt, 2304, 0, 1280 + kk);
#pragma unroll
    for (int i = 0; i < 4; ++i)
      acc[i] = wmma_bf16(a, load_b16(Wv, 1024, nb + 16 * i, kk), acc[i]);
  }
  if (half == 0) {                               // rows m=0..7 = real batch
#pragma unroll
    for (int i = 0; i < 4; ++i)
#pragma unroll
      for (int v = 0; v < 8; ++v)
        out[((size_t)v * 255 + t) * 32000 + nb + 16 * i + n] = acc[i][v];
  }
}

// ===========================================================================
extern "C" void kernel_launch(void* const* d_in, const int* in_sizes, int n_in,
                              void* d_out, int out_size, void* d_ws, size_t ws_size,
                              hipStream_t stream) {
  (void)in_sizes; (void)n_in; (void)out_size; (void)ws_size;
  // setup_inputs() insertion order:
  const float* emb     = (const float*)d_in[0];
  const float* dec_emb = (const float*)d_in[1];
  const float* eW[4][4];                          // [layer*2+dir][Wih,Whh,bih,bhh]
  for (int i = 0; i < 4; ++i)
    for (int j = 0; j < 4; ++j) eW[i][j] = (const float*)d_in[2 + i * 4 + j];
  const float* dWih  = (const float*)d_in[18];
  const float* dWhh  = (const float*)d_in[19];
  const float* dbih  = (const float*)d_in[20];
  const float* dbhh  = (const float*)d_in[21];
  const float* attW1 = (const float*)d_in[22];
  const float* attW2 = (const float*)d_in[23];
  const float* attV  = (const float*)d_in[24];
  const float* Wv    = (const float*)d_in[25];
  const float* bv    = (const float*)d_in[26];
  const int*   x     = (const int*)d_in[27];
  const int*   y     = (const int*)d_in[28];
  float* out = (float*)d_out;

  char* ws = (char*)d_ws;
  size_t off = 0;
  auto alloc = [&](size_t bytes) -> char* {
    off = (off + 255) & ~(size_t)255;
    char* p = ws + off; off += bytes; return p;
  };
  unsigned short* wv_bf = (unsigned short*)alloc(32000ull * 1024 * 2);
  int wih_elems[4] = {2048 * 256, 2048 * 256, 2048 * 1024, 2048 * 1024};
  unsigned short* wih_bf[4];
  for (int i = 0; i < 4; ++i) wih_bf[i] = (unsigned short*)alloc((size_t)wih_elems[i] * 2);
  unsigned short* whh_bf[4];
  for (int i = 0; i < 4; ++i) whh_bf[i] = (unsigned short*)alloc(2048ull * 512 * 2);
  unsigned short* wdec_bf = (unsigned short*)alloc(4096ull * 2304 * 2);
  float*          bsum    = (float*)alloc(4096 * 4);
  unsigned short* xs      = (unsigned short*)alloc(512ull * 16 * 256 * 2);
  unsigned short* l1o     = (unsigned short*)alloc(512ull * 16 * 1024 * 2);
  unsigned short* enco    = (unsigned short*)alloc(512ull * 16 * 1024 * 2);
  float*          G       = (float*)alloc(2ull * 512 * 16 * 2048 * 4);
  float*          proj    = (float*)alloc(8ull * 512 * 10 * 4);
  unsigned short* dins    = (unsigned short*)alloc(255ull * 16 * 256 * 2);
  unsigned short* xt      = (unsigned short*)alloc(16ull * 2304 * 2);
  float*          hdec    = (float*)alloc(16ull * 1024 * 4);
  float*          cdec    = (float*)alloc(16ull * 1024 * 4);
  float*          gdec    = (float*)alloc(16ull * 4096 * 4);

  // one-time (per launch) prep: bf16 conversions, packing, embeds, state init
  k_cvt<<<1024, 256, 0, stream>>>(Wv, wv_bf, 32000ull * 1024);
  for (int i = 0; i < 4; ++i) {
    k_cvt<<<256, 256, 0, stream>>>(eW[i][0], wih_bf[i], (size_t)wih_elems[i]);
    k_cvt<<<256, 256, 0, stream>>>(eW[i][1], whh_bf[i], 2048ull * 512);
  }
  k_pack_wdec<<<1024, 256, 0, stream>>>(dWih, dWhh, wdec_bf);
  k_bsum<<<16, 256, 0, stream>>>(dbih, dbhh, bsum, 4096);
  k_embed_x<<<8192, 256, 0, stream>>>(emb, x, xs);
  k_embed_y<<<4080, 256, 0, stream>>>(dec_emb, y, dins);
  k_zero_state<<<144, 256, 0, stream>>>(xt, hdec, cdec);

  // encoder layer 1 (K=256), then layer 2 (K=1024); G buffer reused
  k_gate_gemm<<<4096, 256, 0, stream>>>(xs, 256, wih_bf[0], wih_bf[1],
                                        eW[0][2], eW[0][3], eW[1][2], eW[1][3], G);
  k_lstm_seq<<<2, 1024, 0, stream>>>(G, whh_bf[0], whh_bf[1], l1o);
  k_gate_gemm<<<4096, 256, 0, stream>>>(l1o, 1024, wih_bf[2], wih_bf[3],
                                        eW[2][2], eW[2][3], eW[3][2], eW[3][3], G);
  k_lstm_seq<<<2, 1024, 0, stream>>>(G, whh_bf[2], whh_bf[3], enco);

  // hoisted W1(enc_out)
  k_encproj<<<512, 256, 0, stream>>>(enco, attW1, proj);

  // decoder: 255 sequential steps
  for (int t = 0; t < 255; ++t) {
    k_att<<<8, 256, 0, stream>>>(t, hdec, attW2, attV, proj, enco, dins, xt);
    k_cell_gemm<<<8, 256, 0, stream>>>(xt, wdec_bf, bsum, gdec);
    k_cell_act<<<16, 1024, 0, stream>>>(gdec, hdec, cdec, xt);
    k_pred<<<63, 256, 0, stream>>>(t, xt, wv_bf, bv, out);
  }
}
